// OmegaRNNMemoryCell_25434796327775
// MI455X (gfx1250) — matrix operand order repaired and checked by reading, not compiled, not run on hardware
//
#include <hip/hip_runtime.h>

#define BB 2
#define TT 1024
#define DD 512
#define HH 8
#define DH 64
#define EE 16
#define CK 4
#define CC 64            // chunk length
#define NC (TT / CC)     // 16 chunks
#define SE 96            // padded extended window (16 back + 64 + 16 pad)
#define SV 80            // valid extended entries

typedef __attribute__((ext_vector_type(16))) _Float16 v16h;
typedef __attribute__((ext_vector_type(8)))  _Float16 v8h;
typedef __attribute__((ext_vector_type(8)))  float    v8f;

__device__ __forceinline__ v16h cat16(v8h lo, v8h hi) {
  return __builtin_shufflevector(lo, hi, 0,1,2,3,4,5,6,7,8,9,10,11,12,13,14,15);
}
// A-fragment (16x32 f16 layout): p = row base + k0, lane half g
__device__ __forceinline__ v16h ld_afrag(const _Float16* p, int g) {
  return cat16(*(const v8h*)(p + 8 * g), *(const v8h*)(p + 16 + 8 * g));
}
// B-fragment (32x16 f16 layout): p = n-row base + k0, lane half g
__device__ __forceinline__ v16h ld_bfrag(const _Float16* p, int g) {
  return cat16(*(const v8h*)(p + 8 * g), *(const v8h*)(p + 16 + 8 * g));
}
__device__ __forceinline__ v16h splat16(_Float16 s) {
  v16h v;
  #pragma unroll
  for (int i = 0; i < 16; ++i) v[i] = s;
  return v;
}
#define WMMA(Aop, Bop, Cop) \
  __builtin_amdgcn_wmma_f32_16x16x32_f16(false, (Aop), false, (Bop), (short)0, (Cop), false, false)

// ---------------------------------------------------------------------------
// Kernel 1: pre-RMSNorm + four per-head sigmoid gates. One block per (b,t).
// ---------------------------------------------------------------------------
__global__ void norm_gates_kernel(const float* __restrict__ x,
                                  const float* __restrict__ w_pre,
                                  const float* __restrict__ w_lr, const float* __restrict__ b_lr,
                                  const float* __restrict__ w_decay, const float* __restrict__ b_decay,
                                  const float* __restrict__ w_mom, const float* __restrict__ b_mom,
                                  const float* __restrict__ w_gate, const float* __restrict__ b_gate,
                                  float* __restrict__ xn,
                                  float* __restrict__ lrA, float* __restrict__ decA,
                                  float* __restrict__ momA, float* __restrict__ uA) {
  __shared__ float sxn[DD];
  __shared__ float red[8];
  __shared__ float stot;
  const int row = blockIdx.x;
  const int b = row / TT, t = row % TT;
  const int tid = threadIdx.x;

  const float a0 = x[(size_t)row * DD + tid];
  const float a1 = x[(size_t)row * DD + tid + 256];
  float ss = a0 * a0 + a1 * a1;
  #pragma unroll
  for (int o = 16; o; o >>= 1) ss += __shfl_xor(ss, o, 32);
  if ((tid & 31) == 0) red[tid >> 5] = ss;
  __syncthreads();
  if (tid == 0) {
    float tot = 0.f;
    #pragma unroll
    for (int i = 0; i < 8; ++i) tot += red[i];
    stot = tot;
  }
  __syncthreads();
  const float inv = rsqrtf(stot * (1.0f / DD) + 1e-6f);
  const float x0 = a0 * inv * w_pre[tid];
  const float x1 = a1 * inv * w_pre[tid + 256];
  xn[(size_t)row * DD + tid] = x0;
  xn[(size_t)row * DD + tid + 256] = x1;
  sxn[tid] = x0;
  sxn[tid + 256] = x1;
  __syncthreads();

  const int grp = tid >> 3, lg = tid & 7;
  const int h = grp >> 2, gsel = grp & 3;
  const float* wp; const float* bp; float* op;
  switch (gsel) {
    case 0: wp = w_lr;    bp = b_lr;    op = lrA;  break;
    case 1: wp = w_decay; bp = b_decay; op = decA; break;
    case 2: wp = w_mom;   bp = b_mom;   op = momA; break;
    default: wp = w_gate; bp = b_gate;  op = uA;   break;
  }
  float acc = 0.f;
  #pragma unroll 8
  for (int i = 0; i < 64; ++i) {
    const int d = lg + 8 * i;
    acc += sxn[d] * wp[d * HH + h];
  }
  acc += __shfl_xor(acc, 1, 32);
  acc += __shfl_xor(acc, 2, 32);
  acc += __shfl_xor(acc, 4, 32);
  if (lg == 0) {
    const float z = acc + bp[h];
    op[((size_t)b * HH + h) * TT + t] = 1.0f / (1.0f + __expf(-z));
  }
}

// ---------------------------------------------------------------------------
// Kernel 2: depthwise conv (K=4, pad 2, truncated) -> f16 GEMM inputs.
// ---------------------------------------------------------------------------
__global__ void dwconv_kernel(const float* __restrict__ xn,
                              const float* __restrict__ cq,
                              const float* __restrict__ ck,
                              const float* __restrict__ cv,
                              _Float16* __restrict__ qh,
                              _Float16* __restrict__ kh,
                              _Float16* __restrict__ vh) {
  const int id = blockIdx.x * blockDim.x + threadIdx.x;
  if (id >= BB * TT * DD) return;
  const int d = id % DD;
  const int row = id / DD;
  const int t = row % TT, b = row / TT;
  float aq = 0.f, ak = 0.f, av = 0.f;
  #pragma unroll
  for (int k = 0; k < CK; ++k) {
    const int ts = t + k - 2;
    if (ts >= 0 && ts < TT) {
      const float v = xn[((size_t)b * TT + ts) * DD + d];
      aq += v * cq[d * CK + k];
      ak += v * ck[d * CK + k];
      av += v * cv[d * CK + k];
    }
  }
  qh[id] = (_Float16)aq;
  kh[id] = (_Float16)ak;
  vh[id] = (_Float16)av;
}

// ---------------------------------------------------------------------------
// Kernel 3: weight transpose + cast (WMMA B operands).
// ---------------------------------------------------------------------------
__global__ void transpose_cast_kernel(const float* __restrict__ W,
                                      _Float16* __restrict__ Wt) {
  const int id = blockIdx.x * blockDim.x + threadIdx.x;
  const int k = id / DD, n = id % DD;
  Wt[(size_t)n * DD + k] = (_Float16)W[(size_t)k * DD + n];
}

// ---------------------------------------------------------------------------
// Kernel 4: WMMA GEMM  C[M,N](f32) = A[M,K](f16 rm) x Bt[N,K](f16 rm),
// with global_prefetch of the upcoming K-block.
// ---------------------------------------------------------------------------
__global__ void wmma_gemm_f16f32(const _Float16* __restrict__ A,
                                 const _Float16* __restrict__ Bt,
                                 float* __restrict__ C,
                                 int M, int N, int K) {
  const int tid = threadIdx.x;
  const int wave = tid >> 5;
  const int lane = tid & 31;
  const int g = lane >> 4;
  const int l15 = lane & 15;
  const int m0 = (blockIdx.y * 8 + wave) * 16;
  const int n0 = blockIdx.x * 16;

  const _Float16* arow = A + (size_t)(m0 + l15) * K;
  const _Float16* brow = Bt + (size_t)(n0 + l15) * K;
  v8f c = {};
  for (int k0 = 0; k0 < K; k0 += 32) {
    __builtin_prefetch(arow + k0 + 256, 0, 3);   // global_prefetch_b8
    __builtin_prefetch(brow + k0 + 256, 0, 3);
    const v16h av = ld_afrag(arow + k0, g);
    const v16h bv = ld_bfrag(brow + k0, g);
    c = WMMA(av, bv, c);
  }
  #pragma unroll
  for (int r = 0; r < 8; ++r)
    C[(size_t)(m0 + r + 8 * g) * N + n0 + l15] = c[r];
}

// ---------------------------------------------------------------------------
// Kernel 5: MultiheadRMSNorm in-place: x / max(||x||/8, 1e-8) * gamma.
// ---------------------------------------------------------------------------
__global__ void mh_rmsnorm_kernel(float* __restrict__ Qf,
                                  const float* __restrict__ gamma) {
  const int idx = blockIdx.x;
  const int h = idx % HH;
  const int row = idx / HH;
  const int j = threadIdx.x;
  float* base = Qf + (size_t)row * (HH * DH) + h * DH;
  const float v0 = base[j];
  const float v1 = base[j + 32];
  float ss = v0 * v0 + v1 * v1;
  #pragma unroll
  for (int o = 16; o; o >>= 1) ss += __shfl_xor(ss, o, 32);
  const float denom = fmaxf(sqrtf(ss) * 0.125f, 1e-8f);
  const float inv = 1.0f / denom;
  base[j]      = v0 * inv * gamma[h * DH + j];
  base[j + 32] = v1 * inv * gamma[h * DH + j + 32];
}

// ---------------------------------------------------------------------------
// Kernel 6: per-chunk scalar coefficients (see round-2 derivation).
// ---------------------------------------------------------------------------
__global__ void coef_kernel(const float* __restrict__ lrA, const float* __restrict__ decA,
                            const float* __restrict__ momA, const float* __restrict__ uA,
                            float* __restrict__ Pmask, float* __restrict__ alphaA,
                            float* __restrict__ betaA, float* __restrict__ dZA,
                            float* __restrict__ pmAllA) {
  __shared__ float wlds[CC];
  __shared__ float m_s[CC], a_s[CC], l_s[CC], u_e[SE];
  const int bhc = blockIdx.x;
  const int bh = bhc / NC, cidx = bhc % NC;
  const int t0 = cidx * CC;
  const int tid = threadIdx.x;
  const size_t gbase = (size_t)bh * TT;
  if (tid < CC) {
    m_s[tid] = momA[gbase + t0 + tid];
    a_s[tid] = decA[gbase + t0 + tid];
    l_s[tid] = lrA[gbase + t0 + tid];
  }
  if (tid < SE) {
    const int s_abs = t0 - EE + tid;
    u_e[tid] = (tid < SV && s_abs >= 0) ? uA[gbase + s_abs] : 0.f;
  }
  __syncthreads();

  float Pm = 0.f, W = 0.f;
  float al = 1.f, be = 0.f, pmAll = 1.f;
  for (int t = 0; t < CC; ++t) {
    const float mt = m_s[t], at = a_s[t];
    if (tid < CC) {
      if (t == tid)      { Pm = 1.f; W = 1.f; }
      else if (t > tid)  { Pm *= mt; W = at * W + Pm; }
      wlds[tid] = l_s[tid] * W;
    }
    al *= at;
    pmAll *= mt;
    be = at * be + pmAll;
    if (tid == 0) {
      alphaA[(size_t)bhc * CC + t] = al;
      betaA[(size_t)bhc * CC + t]  = be;
    }
    __syncthreads();
    if (tid < SE) {
      const int s_rel = tid - EE;
      int lo = s_rel > 0 ? s_rel : 0;
      int hi = s_rel + EE - 1; if (hi > t) hi = t;
      float cacc = 0.f;
      for (int p = lo; p <= hi; ++p) cacc += wlds[p];
      Pmask[((size_t)bhc * CC + t) * SE + tid] = u_e[tid] * cacc;
    }
    __syncthreads();
  }
  if (tid < CC) wlds[tid] = l_s[tid] * Pm;   // Pm == Pm(e, p')
  __syncthreads();
  if (tid < SE) {
    const int s_rel = tid - EE;
    int lo = s_rel > 0 ? s_rel : 0;
    int hi = s_rel + EE - 1; if (hi > CC - 1) hi = CC - 1;
    float cacc = 0.f;
    for (int p = lo; p <= hi; ++p) cacc += wlds[p];
    dZA[(size_t)bhc * SE + tid] = u_e[tid] * cacc;
  }
  if (tid == 0) pmAllA[bhc] = pmAll;
}

// ---------------------------------------------------------------------------
// Kernel 7: chunkwise recurrence on WMMA, software-pipelined: chunk c+1's
// global loads are issued into registers right after B1 of chunk c, so their
// HBM latency overlaps the WMMA phases; the register->LDS stage at the top
// of the next iteration is the only consumer.
// ---------------------------------------------------------------------------
__global__ void chunk_kernel(const float* __restrict__ Qf, const float* __restrict__ Kf,
                             const float* __restrict__ Vf,
                             const float* __restrict__ Pmask, const float* __restrict__ alphaA,
                             const float* __restrict__ betaA, const float* __restrict__ dZA,
                             const float* __restrict__ pmAllA,
                             _Float16* __restrict__ Oh) {
  __shared__ alignas(16) _Float16 Qh[CC][DH];
  __shared__ alignas(16) _Float16 Kh[SE][DH];
  __shared__ alignas(16) _Float16 KhT[DH][SE];
  __shared__ alignas(16) _Float16 VhT[DH][SE];
  __shared__ alignas(16) _Float16 PAh[CC][SE];
  __shared__ alignas(16) _Float16 Sh[DH][DH];
  __shared__ alignas(16) _Float16 Zh[DH][DH];
  __shared__ alignas(16) _Float16 dS16[SE];
  __shared__ alignas(16) _Float16 dZ16[SE];
  __shared__ float alC[CC], beC[CC];

  const int bh = blockIdx.x;
  const int b = bh / HH, h = bh % HH;
  const int tid = threadIdx.x;
  const int wave = tid >> 5, lane = tid & 31;
  const int g = lane >> 4, l15 = lane & 15;

  for (int i = tid; i < DH * DH; i += 256) {
    Sh[i / DH][i % DH] = (_Float16)0.f;
    Zh[i / DH][i % DH] = (_Float16)0.f;
  }
  float Sreg[2][8], Zreg[2][8];
  #pragma unroll
  for (int w = 0; w < 2; ++w)
    #pragma unroll
    for (int r = 0; r < 8; ++r) { Sreg[w][r] = 0.f; Zreg[w][r] = 0.f; }

  // ---- double-buffer registers for the next chunk's operands
  float qbuf[16], kbuf[24], vbuf[24];
  float albuf = 0.f, bebuf = 0.f, dsbuf = 0.f, dzbuf = 0.f, pmabuf = 1.f;

  auto load_chunk = [&](int cc) {
    const int bhcN = bh * NC + cc;
    const int t0N = cc * CC;
    if (tid < CC) {
      albuf = alphaA[(size_t)bhcN * CC + tid];
      bebuf = betaA[(size_t)bhcN * CC + tid];
    }
    if (tid < SE) {
      dsbuf = Pmask[((size_t)bhcN * CC + (CC - 1)) * SE + tid];
      dzbuf = dZA[(size_t)bhcN * SE + tid];
    }
    pmabuf = pmAllA[bhcN];
    #pragma unroll
    for (int it = 0; it < 16; ++it) {
      const int i = tid + it * 256;
      const int tr = i >> 6, j = i & 63;
      qbuf[it] = Qf[((size_t)b * TT + t0N + tr) * (HH * DH) + h * DH + j];
    }
    #pragma unroll
    for (int it = 0; it < 24; ++it) {
      const int i = tid + it * 256;
      const int s = i >> 6, j = i & 63;
      const int s_abs = t0N - EE + s;
      float kv = 0.f, vv = 0.f;
      if (s < SV && s_abs >= 0) {
        const size_t rb = ((size_t)b * TT + s_abs) * (HH * DH) + h * DH;
        kv = Kf[rb + j];
        vv = Vf[rb + j];
      }
      kbuf[it] = kv;
      vbuf[it] = vv;
    }
    // prefetch the Pmask block consumed by phase 1 of chunk cc
    const char* pmBase = (const char*)(Pmask + (size_t)bhcN * CC * SE);
    if (tid < 192) __builtin_prefetch(pmBase + tid * 128, 0, 3);
  };

  load_chunk(0);

  for (int c = 0; c < NC; ++c) {
    const int bhc = bh * NC + c;
    const int t0 = c * CC;
    // ---- phase 0: registers -> LDS (convert + transposes)
    if (tid < CC) { alC[tid] = albuf; beC[tid] = bebuf; }
    if (tid < SE) { dS16[tid] = (_Float16)dsbuf; dZ16[tid] = (_Float16)dzbuf; }
    const float pma = pmabuf;
    #pragma unroll
    for (int it = 0; it < 16; ++it) {
      const int i = tid + it * 256;
      Qh[i >> 6][i & 63] = (_Float16)qbuf[it];
    }
    #pragma unroll
    for (int it = 0; it < 24; ++it) {
      const int i = tid + it * 256;
      const int s = i >> 6, j = i & 63;
      const _Float16 kh16 = (_Float16)kbuf[it];
      Kh[s][j]  = kh16;
      KhT[j][s] = kh16;
      VhT[j][s] = (_Float16)vbuf[it];
    }
    __syncthreads();  // B1

    // ---- issue next chunk's global loads (latency hides behind ph1..ph3)
    if (c + 1 < NC) load_chunk(c + 1);

    // ---- phase 1: masked scores -> PAh
    const float* Prow = Pmask + (size_t)bhc * CC * SE;
    #pragma unroll
    for (int w = 0; w < 3; ++w) {
      const int tile = wave * 3 + w;           // 24 tiles (4 x 6)
      const int tm = tile / 6, tn = tile % 6;
      v8f acc = {};
      #pragma unroll
      for (int k0 = 0; k0 < DH; k0 += 32) {
        const v16h av = ld_afrag(&Qh[tm * 16 + l15][k0], g);
        const v16h bv = ld_bfrag(&Kh[tn * 16 + l15][k0], g);
        acc = WMMA(av, bv, acc);
      }
      #pragma unroll
      for (int r = 0; r < 8; ++r) {
        const int rt = tm * 16 + r + 8 * g, cs = tn * 16 + l15;
        PAh[rt][cs] = (_Float16)(acc[r] * Prow[(size_t)rt * SE + cs]);
      }
    }
    __syncthreads();  // B2

    // ---- phase 2: y = diag(al) Q S^T + diag(be) Q Z^T + (P o A) V
    #pragma unroll
    for (int w = 0; w < 2; ++w) {
      const int tile = wave * 2 + w;           // 16 tiles (4 x 4)
      const int tm = tile >> 2, tn = tile & 3;
      const int rowA = tm * 16 + l15;
      const v16h asc = splat16((_Float16)alC[rowA]);
      const v16h bsc = splat16((_Float16)beC[rowA]);
      v8f acc = {};
      #pragma unroll
      for (int k0 = 0; k0 < DH; k0 += 32) {
        const v16h qv = ld_afrag(&Qh[rowA][k0], g);
        const v16h bs = ld_bfrag(&Sh[tn * 16 + l15][k0], g);
        const v16h bz = ld_bfrag(&Zh[tn * 16 + l15][k0], g);
        acc = WMMA(qv * asc, bs, acc);
        acc = WMMA(qv * bsc, bz, acc);
      }
      #pragma unroll
      for (int k0 = 0; k0 < SE; k0 += 32) {
        const v16h av = ld_afrag(&PAh[rowA][k0], g);
        const v16h bv = ld_bfrag(&VhT[tn * 16 + l15][k0], g);
        acc = WMMA(av, bv, acc);
      }
      #pragma unroll
      for (int r = 0; r < 8; ++r) {
        const int rt = tm * 16 + r + 8 * g, ci = tn * 16 + l15;
        Oh[((size_t)b * TT + t0 + rt) * (HH * DH) + h * DH + ci] = (_Float16)acc[r];
      }
    }

    // ---- phase 3: state-update GEMMs (read only VhT/KhT/dS16/dZ16)
    v8f sacc[2], zacc[2];
    #pragma unroll
    for (int w = 0; w < 2; ++w) {
      const int pos = wave * 2 + w;            // 16 positions (4 x 4)
      const int ti = pos >> 2, tj = pos & 3;
      v8f cs = {}, cz = {};
      #pragma unroll
      for (int k0 = 0; k0 < SE; k0 += 32) {
        const v16h vf = ld_afrag(&VhT[ti * 16 + l15][k0], g);
        const v16h ds = ld_afrag(&dS16[k0], g);
        const v16h dz = ld_afrag(&dZ16[k0], g);
        const v16h bv = ld_bfrag(&KhT[tj * 16 + l15][k0], g);
        cs = WMMA(vf * ds, bv, cs);
        cz = WMMA(vf * dz, bv, cz);
      }
      sacc[w] = cs; zacc[w] = cz;
    }
    const float ae = alC[CC - 1], bee = beC[CC - 1];
    __syncthreads();  // B3: everyone done reading Sh/Zh in phase 2
    #pragma unroll
    for (int w = 0; w < 2; ++w) {
      const int pos = wave * 2 + w;
      const int ti = pos >> 2, tj = pos & 3;
      #pragma unroll
      for (int r = 0; r < 8; ++r) {
        const float so = Sreg[w][r], zo = Zreg[w][r];
        const float zn = pma * zo + zacc[w][r];
        const float sn = ae * so + bee * zo + sacc[w][r];
        Sreg[w][r] = sn; Zreg[w][r] = zn;
        const int ri = ti * 16 + r + 8 * g, cj = tj * 16 + l15;
        Sh[ri][cj] = (_Float16)sn;
        Zh[ri][cj] = (_Float16)zn;
      }
    }
    __syncthreads();  // B4
  }
}

// ---------------------------------------------------------------------------
extern "C" void kernel_launch(void* const* d_in, const int* in_sizes, int n_in,
                              void* d_out, int out_size, void* d_ws, size_t ws_size,
                              hipStream_t stream) {
  (void)in_sizes; (void)n_in; (void)out_size; (void)ws_size;
  const float* x       = (const float*)d_in[0];
  const float* w_pre   = (const float*)d_in[1];
  const float* conv_q  = (const float*)d_in[2];
  const float* conv_k  = (const float*)d_in[3];
  const float* conv_v  = (const float*)d_in[4];
  const float* Wq      = (const float*)d_in[5];
  const float* Wk      = (const float*)d_in[6];
  const float* Wv      = (const float*)d_in[7];
  const float* Wo      = (const float*)d_in[8];
  const float* w_lr    = (const float*)d_in[9];
  const float* b_lr    = (const float*)d_in[10];
  const float* w_decay = (const float*)d_in[11];
  const float* b_decay = (const float*)d_in[12];
  const float* w_mom   = (const float*)d_in[13];
  const float* b_mom   = (const float*)d_in[14];
  const float* w_gate  = (const float*)d_in[15];
  const float* b_gate  = (const float*)d_in[16];
  const float* gamma_q = (const float*)d_in[17];
  const float* gamma_k = (const float*)d_in[18];
  float* out = (float*)d_out;

  const size_t M = (size_t)BB * TT;           // 2048
  const int NBH = BB * HH;                    // 16
  char* p = (char*)d_ws;
  auto alloc = [&](size_t bytes) -> void* {
    void* r = (void*)p;
    p += (bytes + 255) & ~(size_t)255;
    return r;
  };
  float*    xn     = (float*)alloc(M * DD * 4);
  float*    lrA    = (float*)alloc((size_t)NBH * TT * 4);
  float*    decA   = (float*)alloc((size_t)NBH * TT * 4);
  float*    momA   = (float*)alloc((size_t)NBH * TT * 4);
  float*    uA     = (float*)alloc((size_t)NBH * TT * 4);
  _Float16* qinH   = (_Float16*)alloc(M * DD * 2);
  _Float16* kinH   = (_Float16*)alloc(M * DD * 2);
  _Float16* vinH   = (_Float16*)alloc(M * DD * 2);
  _Float16* WqT    = (_Float16*)alloc((size_t)DD * DD * 2);
  _Float16* WkT    = (_Float16*)alloc((size_t)DD * DD * 2);
  _Float16* WvT    = (_Float16*)alloc((size_t)DD * DD * 2);
  _Float16* WoT    = (_Float16*)alloc((size_t)DD * DD * 2);
  float*    Qf     = (float*)alloc(M * DD * 4);
  float*    Kf     = (float*)alloc(M * DD * 4);
  float*    Vf     = (float*)alloc(M * DD * 4);
  _Float16* Oh     = (_Float16*)alloc(M * DD * 2);
  float*    Pmask  = (float*)alloc((size_t)NBH * NC * CC * SE * 4);
  float*    alphaA = (float*)alloc((size_t)NBH * NC * CC * 4);
  float*    betaA  = (float*)alloc((size_t)NBH * NC * CC * 4);
  float*    dZA    = (float*)alloc((size_t)NBH * NC * SE * 4);
  float*    pmAllA = (float*)alloc((size_t)NBH * NC * 4);

  norm_gates_kernel<<<(int)M, 256, 0, stream>>>(x, w_pre, w_lr, b_lr, w_decay, b_decay,
                                                w_mom, b_mom, w_gate, b_gate,
                                                xn, lrA, decA, momA, uA);
  dwconv_kernel<<<(int)((M * DD + 255) / 256), 256, 0, stream>>>(xn, conv_q, conv_k, conv_v,
                                                                 qinH, kinH, vinH);
  const int wtBlocks = (DD * DD + 255) / 256;
  transpose_cast_kernel<<<wtBlocks, 256, 0, stream>>>(Wq, WqT);
  transpose_cast_kernel<<<wtBlocks, 256, 0, stream>>>(Wk, WkT);
  transpose_cast_kernel<<<wtBlocks, 256, 0, stream>>>(Wv, WvT);
  transpose_cast_kernel<<<wtBlocks, 256, 0, stream>>>(Wo, WoT);
  dim3 ggrid(DD / 16, (unsigned)(M / (16 * 8)));
  wmma_gemm_f16f32<<<ggrid, 256, 0, stream>>>(qinH, WqT, Qf, (int)M, DD, DD);
  wmma_gemm_f16f32<<<ggrid, 256, 0, stream>>>(kinH, WkT, Kf, (int)M, DD, DD);
  wmma_gemm_f16f32<<<ggrid, 256, 0, stream>>>(vinH, WvT, Vf, (int)M, DD, DD);
  mh_rmsnorm_kernel<<<(int)(M * HH), 32, 0, stream>>>(Qf, gamma_q);
  mh_rmsnorm_kernel<<<(int)(M * HH), 32, 0, stream>>>(Kf, gamma_k);
  coef_kernel<<<NBH * NC, 128, 0, stream>>>(lrA, decA, momA, uA,
                                            Pmask, alphaA, betaA, dZA, pmAllA);
  chunk_kernel<<<NBH, 256, 0, stream>>>(Qf, Kf, Vf, Pmask, alphaA, betaA, dZA, pmAllA, Oh);
  wmma_gemm_f16f32<<<ggrid, 256, 0, stream>>>(Oh, WoT, out, (int)M, DD, DD);
}
